// Informer_27590869909768
// MI455X (gfx1250) — matrix operand order repaired and checked by reading, not compiled, not run on hardware
//
#include <hip/hip_runtime.h>
#include <math.h>

// ---------------- constants ----------------
#define DM      512
#define DFF     512
#define HEADS   8
#define HD      64
#define SEQL    512
#define BATCHN  16
#define BH      (BATCHN*HEADS)      // 128
#define BLD     (BATCHN*SEQL)       // 8192 rows
#define U_SAMP  35                  // 5*ceil(ln 512)
#define GN      512                 // dense GEMM N (all call sites)
#define GK      512                 // dense GEMM K (all call sites)
static constexpr size_t T = (size_t)BATCHN * SEQL * DM;   // 4,194,304 floats
static constexpr size_t FDIM = (size_t)DM * SEQL;          // 262144

typedef __bf16 bhalf;
typedef __attribute__((ext_vector_type(16))) __bf16 v16bf;
typedef __attribute__((ext_vector_type(8)))  __bf16 v8bf;
typedef __attribute__((ext_vector_type(4)))  __bf16 v4bf;
typedef __attribute__((ext_vector_type(8)))  float  v8f;

__device__ __forceinline__ v8f wmma_bf16(v16bf a, v16bf b, v8f c) {
  // D = A(16x32 bf16) * B(32x16 bf16) + C(16x16 f32)
  return __builtin_amdgcn_wmma_f32_16x16x32_bf16(false, a, false, b, (short)0, c,
                                                 false, false);
}
// assemble a 16-element fragment from two 16B-aligned LDS vectors (2x ds_load_b128)
__device__ __forceinline__ v16bf ld_frag(const bhalf* p0, const bhalf* p1) {
  v8bf a = *(const v8bf*)p0;
  v8bf b = *(const v8bf*)p1;
  return __builtin_shufflevector(a, b, 0,1,2,3,4,5,6,7,8,9,10,11,12,13,14,15);
}
__device__ __forceinline__ v4bf cvt4(float4 v) {
  v4bf o; o[0] = (bhalf)v.x; o[1] = (bhalf)v.y; o[2] = (bhalf)v.z; o[3] = (bhalf)v.w;
  return o;
}
__device__ __forceinline__ float gelu_f(float x) {
  return 0.5f * x * (1.f + erff(x * 0.70710678118654752f));
}
__device__ __forceinline__ unsigned hash_u(unsigned x) {
  x ^= x >> 16; x *= 0x7feb352dU; x ^= x >> 15; x *= 0x846ca68bU; x ^= x >> 16;
  return x;
}

// ======================================================================
// Dense GEMM: C[M,512] = act(A[M,512] @ B[512,512] + bias)  (bf16 WMMA)
// BM=128 BN=64 BK=32, 256 threads = 8 wave32 waves, wave tile 32x32 (2x2 WMMA)
// Double-buffered LDS staging: global loads for tile k+1 issued before the
// barrier, WMMA on tile k, then regs -> LDS buf^1. One barrier per K-step.
// REQUIRES: M%128==0.
// ======================================================================
__global__ __launch_bounds__(256) void gemm_wmma_kernel(
    const float* __restrict__ A, const float* __restrict__ B,
    const float* __restrict__ bias, float* __restrict__ C,
    int M, int act)
{
  // row strides are 16B multiples so fragments load as 2x ds_load_b128
  __shared__ alignas(16) bhalf As[2][128][40];   // [M][K], 80B rows
  __shared__ alignas(16) bhalf Bs[2][64][40];    // [N][K] (transposed), 80B rows

  const int tid = threadIdx.x;
  const int lane = tid & 31;
  const int w = tid >> 5;
  const int g = lane >> 4;       // half-wave group
  const int lr = lane & 15;
  const int m0 = blockIdx.y * 128;
  const int n0 = blockIdx.x * 64;
  const int mo = (w >> 1) * 32;  // wave M offset in block tile
  const int no = (w & 1) * 32;   // wave N offset

  float4 areg[4];
  float  breg[8];

  // A: thread owns 4 float4 slots; B: thread owns 2 (n, k-quad) tasks
  auto load_tile = [&](int k0) {
    #pragma unroll
    for (int t = 0; t < 4; ++t) {
      int i = tid + t * 256; int r = i >> 3, c4 = (i & 7) * 4;
      areg[t] = *(const float4*)(A + (size_t)(m0 + r) * GK + k0 + c4);
    }
    const float* bp = B + (size_t)k0 * GN + n0;    // rows are immediate offsets
    #pragma unroll
    for (int t = 0; t < 2; ++t) {
      int i = tid + t * 256; int n = i & 63, kq = (i >> 6) * 4;
      #pragma unroll
      for (int j = 0; j < 4; ++j)
        breg[t * 4 + j] = bp[(size_t)(kq + j) * GN + n];
    }
  };
  auto store_tile = [&](int bi) {
    #pragma unroll
    for (int t = 0; t < 4; ++t) {
      int i = tid + t * 256; int r = i >> 3, c4 = (i & 7) * 4;
      *(v4bf*)&As[bi][r][c4] = cvt4(areg[t]);
    }
    #pragma unroll
    for (int t = 0; t < 2; ++t) {
      int i = tid + t * 256; int n = i & 63, kq = (i >> 6) * 4;
      v4bf o;
      #pragma unroll
      for (int j = 0; j < 4; ++j) o[j] = (bhalf)breg[t * 4 + j];
      *(v4bf*)&Bs[bi][n][kq] = o;
    }
  };

  v8f acc[2][2];
  #pragma unroll
  for (int i = 0; i < 2; ++i)
    #pragma unroll
    for (int j = 0; j < 2; ++j)
      acc[i][j] = (v8f){0.f,0.f,0.f,0.f,0.f,0.f,0.f,0.f};

  load_tile(0);
  store_tile(0);
  int cur = 0;

  for (int k0 = 0; k0 < GK; k0 += 32) {
    const bool more = (k0 + 32 < GK);
    if (more) load_tile(k0 + 32);                   // global loads in flight
    if (k0 + 64 < GK)                               // two tiles ahead -> L2
      __builtin_prefetch(A + (size_t)(m0 + (tid >> 3)) * GK + k0 + 64 + (tid & 7) * 4, 0, 1);
    __syncthreads();                                // tile[cur] visible

    v16bf af[2], bfr[2];
    #pragma unroll
    for (int mt = 0; mt < 2; ++mt) {
      const bhalf* ap = &As[cur][mo + mt * 16 + lr][0];
      af[mt] = ld_frag(ap + g * 8, ap + 16 + g * 8);   // K = g*8+e | 16+g*8+e
    }
    #pragma unroll
    for (int nt = 0; nt < 2; ++nt) {
      const bhalf* bp2 = &Bs[cur][no + nt * 16 + lr][0];
      bfr[nt] = ld_frag(bp2 + g * 8, bp2 + 16 + g * 8); // K = g*8+e | 16+g*8+e
    }
    #pragma unroll
    for (int mt = 0; mt < 2; ++mt)
      #pragma unroll
      for (int nt = 0; nt < 2; ++nt)
        acc[mt][nt] = wmma_bf16(af[mt], bfr[nt], acc[mt][nt]);

    if (more) store_tile(cur ^ 1);                  // fill other buffer
    cur ^= 1;
  }

  // ---- epilogue: bias + activation
  #pragma unroll
  for (int mt = 0; mt < 2; ++mt)
    #pragma unroll
    for (int nt = 0; nt < 2; ++nt) {
      int n = n0 + no + nt * 16 + lr;
      float bv = bias ? bias[n] : 0.f;
      #pragma unroll
      for (int r = 0; r < 8; ++r) {
        int m = m0 + mo + mt * 16 + g * 8 + r;   // C/D: M = r + g*8
        float v = acc[mt][nt][r] + bv;
        if (act == 1) v = gelu_f(v);
        else if (act == 2) v = fmaxf(v, 0.f);
        C[(size_t)m * GN + n] = v;
      }
    }
}

// ======================================================================
// Fused full attention (flash-style, scores never leave LDS)
// Q:[B,SEQL,H,HD]  K,V:[B,S,H,HD]  O:[B,SEQL,H,HD]   (S % 64 == 0)
// block = 64 query rows, 8 waves; S streamed in 64-key chunks; online softmax
// ======================================================================
__global__ __launch_bounds__(256) void attn_fused_kernel(
    const float* __restrict__ Q, const float* __restrict__ Kg,
    const float* __restrict__ Vg, float* __restrict__ O, int S, float scale)
{
  __shared__ alignas(16) bhalf Qs[64][80];  // [q][d], 160B rows
  __shared__ alignas(16) bhalf Ks[64][80];  // [s][d]
  __shared__ alignas(16) bhalf Vt[64][80];  // [d][s]  (transposed for PV B-frags)
  __shared__ alignas(16) bhalf Pb[64][80];  // [q][s]  softmax probs as bf16
  __shared__ float Sc[64][68];              // raw scores fp32
  __shared__ float Ml[64], Ll[64], Al[64];

  const int tid = threadIdx.x, lane = tid & 31, w = tid >> 5;
  const int g = lane >> 4, lr = lane & 15;
  const int l0 = blockIdx.x * 64;
  const int b = blockIdx.y / HEADS, h = blockIdx.y % HEADS;
  const int mo = (w >> 1) * 16;   // wave: 16 rows x 32 cols
  const int no = (w & 1) * 32;

  // stage Q block (arena buffers are 16B aligned)
  #pragma unroll
  for (int t = 0; t < 4; ++t) {
    int i = tid + t * 256; int r = i >> 4, c4 = (i & 15) * 4;
    float4 v = *(const float4*)(Q + (((size_t)b * SEQL + l0 + r) * HEADS + h) * HD + c4);
    *(v4bf*)&Qs[r][c4] = cvt4(v);
  }
  if (tid < 64) { Ml[tid] = -1e30f; Ll[tid] = 0.f; }
  __syncthreads();

  v8f oacc[2];
  oacc[0] = (v8f){0.f,0.f,0.f,0.f,0.f,0.f,0.f,0.f};
  oacc[1] = oacc[0];

  for (int s0 = 0; s0 < S; s0 += 64) {
    // ---- stage K chunk [s][d] and V chunk transposed [d][s]
    // issue all 8 global loads first, then convert/store
    float4 kreg[4], vreg[4];
    #pragma unroll
    for (int t = 0; t < 4; ++t) {
      int i = tid + t * 256; int r = i >> 4, c4 = (i & 15) * 4;
      kreg[t] = *(const float4*)(Kg + (((size_t)b * S + s0 + r) * HEADS + h) * HD + c4);
      vreg[t] = *(const float4*)(Vg + (((size_t)b * S + s0 + r) * HEADS + h) * HD + c4);
    }
    #pragma unroll
    for (int t = 0; t < 4; ++t) {
      int i = tid + t * 256; int r = i >> 4, c4 = (i & 15) * 4;
      *(v4bf*)&Ks[r][c4] = cvt4(kreg[t]);
      Vt[c4 + 0][r] = (bhalf)vreg[t].x; Vt[c4 + 1][r] = (bhalf)vreg[t].y;
      Vt[c4 + 2][r] = (bhalf)vreg[t].z; Vt[c4 + 3][r] = (bhalf)vreg[t].w;
    }
    __syncthreads();

    // ---- scores 64x64 = Qs @ Ks^T * scale
    v8f sacc[2];
    sacc[0] = (v8f){0.f,0.f,0.f,0.f,0.f,0.f,0.f,0.f};
    sacc[1] = sacc[0];
    #pragma unroll
    for (int kk = 0; kk < 64; kk += 32) {
      const bhalf* qp = &Qs[mo + lr][kk];
      v16bf aq = ld_frag(qp + g * 8, qp + 16 + g * 8);
      #pragma unroll
      for (int nt = 0; nt < 2; ++nt) {
        const bhalf* kp = &Ks[no + nt * 16 + lr][kk];   // B[kd][s] = Ks[s][kd]
        v16bf bk = ld_frag(kp + g * 8, kp + 16 + g * 8);
        sacc[nt] = wmma_bf16(aq, bk, sacc[nt]);
      }
    }
    #pragma unroll
    for (int nt = 0; nt < 2; ++nt) {
      int col = no + nt * 16 + lr;
      #pragma unroll
      for (int r = 0; r < 8; ++r)
        Sc[mo + g * 8 + r][col] = sacc[nt][r] * scale;
    }
    __syncthreads();

    // ---- online softmax: wave w owns rows 8w..8w+7; probs stored bf16 in Pb
    for (int rr = 0; rr < 8; ++rr) {
      int row = w * 8 + rr;
      float v0 = Sc[row][lane], v1 = Sc[row][lane + 32];
      float mx = fmaxf(v0, v1);
      #pragma unroll
      for (int off = 16; off >= 1; off >>= 1) mx = fmaxf(mx, __shfl_xor(mx, off, 32));
      float mnew = fmaxf(Ml[row], mx);
      float p0 = expf(v0 - mnew), p1 = expf(v1 - mnew);
      Pb[row][lane] = (bhalf)p0; Pb[row][lane + 32] = (bhalf)p1;
      float s = p0 + p1;
      #pragma unroll
      for (int off = 16; off >= 1; off >>= 1) s += __shfl_xor(s, off, 32);
      if (lane == 0) {
        float alpha = expf(Ml[row] - mnew);
        Al[row] = alpha;
        Ll[row] = Ll[row] * alpha + s;
        Ml[row] = mnew;
      }
    }
    __syncthreads();

    // ---- rescale O accumulators by alpha of their rows
    #pragma unroll
    for (int nt = 0; nt < 2; ++nt)
      #pragma unroll
      for (int r = 0; r < 8; ++r)
        oacc[nt][r] *= Al[mo + g * 8 + r];

    // ---- O += P(64x64) @ V(64x64): A from Pb, B from Vt (both K-contiguous)
    #pragma unroll
    for (int kk = 0; kk < 64; kk += 32) {
      const bhalf* pp = &Pb[mo + lr][kk];
      v16bf ap = ld_frag(pp + g * 8, pp + 16 + g * 8);
      #pragma unroll
      for (int nt = 0; nt < 2; ++nt) {
        const bhalf* vp = &Vt[no + nt * 16 + lr][kk];   // B[s][d] = Vt[d][s]
        v16bf bv = ld_frag(vp + g * 8, vp + 16 + g * 8);
        oacc[nt] = wmma_bf16(ap, bv, oacc[nt]);
      }
    }
    __syncthreads();
  }

  // ---- epilogue: normalize by running sum, write [B,L,H,HD]
  #pragma unroll
  for (int nt = 0; nt < 2; ++nt) {
    int d = no + nt * 16 + lr;
    #pragma unroll
    for (int r = 0; r < 8; ++r) {
      int row = mo + g * 8 + r;
      O[(((size_t)b * SEQL + l0 + row) * HEADS + h) * HD + d] = oacc[nt][r] / Ll[row];
    }
  }
}

// ======================================================================
// Memory-bound helper kernels (VALU; bandwidth-limited, WMMA irrelevant)
// ======================================================================

// out = LN(x [+ res]) * g + b ; row length DM, one block per row, in-place safe
__global__ __launch_bounds__(256) void ln_kernel(
    const float* __restrict__ x, const float* __restrict__ res,
    const float* __restrict__ gam, const float* __restrict__ bet,
    float* __restrict__ out)
{
  __shared__ float s1[256], s2[256];
  const int tid = threadIdx.x;
  const size_t row = blockIdx.x;
  const float* xr = x + row * DM;
  float v0 = xr[tid], v1 = xr[tid + 256];
  if (res) { v0 += res[row * DM + tid]; v1 += res[row * DM + tid + 256]; }
  s1[tid] = v0 + v1; s2[tid] = v0 * v0 + v1 * v1;
  __syncthreads();
  for (int s = 128; s > 0; s >>= 1) {
    if (tid < s) { s1[tid] += s1[tid + s]; s2[tid] += s2[tid + s]; }
    __syncthreads();
  }
  float mean = s1[0] * (1.f / DM);
  float var = s2[0] * (1.f / DM) - mean * mean;
  float inv = rsqrtf(var + 1e-5f);
  out[row * DM + tid]       = (v0 - mean) * inv * gam[tid]       + bet[tid];
  out[row * DM + tid + 256] = (v1 - mean) * inv * gam[tid + 256] + bet[tid + 256];
}

// conv1d(kernel 3, wrap pad) token embedding + sinusoidal position embedding
__global__ void embed_kernel(const float* __restrict__ x, const float* __restrict__ W,
                             float* __restrict__ out)
{
  size_t idx = (size_t)blockIdx.x * 256 + threadIdx.x;
  if (idx >= T) return;
  int d = idx % DM; int l = (idx / DM) % SEQL; int b = idx / ((size_t)DM * SEQL);
  float acc = 0.f;
  #pragma unroll
  for (int c = 0; c < 3; ++c)
    #pragma unroll
    for (int t = 0; t < 3; ++t) {
      int ls = (l - 1 + t + SEQL) % SEQL;
      acc += x[((size_t)b * SEQL + ls) * 3 + c] * W[(d * 3 + c) * 3 + t];
    }
  int p = d >> 1;
  float freq = expf(-(float)(2 * p) * (logf(10000.f) / DM));
  float ang = (float)l * freq;
  acc += (d & 1) ? cosf(ang) : sinf(ang);
  out[idx] = acc;
}

// enc_mix[b,l,d] = enc_loc[b,l,d] * softmax_over_l(enc_arr[b,:,d])
__global__ void mix_softmax_kernel(const float* __restrict__ arr,
                                   const float* __restrict__ loc,
                                   float* __restrict__ out)
{
  int idx = blockIdx.x * 256 + threadIdx.x;          // over B*DM columns
  if (idx >= BATCHN * DM) return;
  int b = idx / DM, d = idx % DM;
  const float* base = arr + (size_t)b * SEQL * DM + d;
  float mx = -1e30f;
  for (int l = 0; l < SEQL; ++l) mx = fmaxf(mx, base[(size_t)l * DM]);
  float s = 0.f;
  for (int l = 0; l < SEQL; ++l) s += expf(base[(size_t)l * DM] - mx);
  float inv = 1.f / s;
  for (int l = 0; l < SEQL; ++l) {
    size_t o = (size_t)b * SEQL * DM + (size_t)l * DM + d;
    out[o] = loc[o] * expf(base[(size_t)l * DM] - mx) * inv;
  }
}

// ---- prob-sparse attention helpers (u = U = 35; tiny compute) ----
__global__ void prob_sample_kernel(const float* __restrict__ Q,
                                   const float* __restrict__ K,
                                   float* __restrict__ Msc, unsigned rng)
{
  int idx = blockIdx.x * 256 + threadIdx.x;          // (b,h,l)
  if (idx >= BATCHN * HEADS * SEQL) return;
  int l = idx % SEQL; int h = (idx / SEQL) % HEADS; int b = idx / (SEQL * HEADS);
  const float* q = Q + (((size_t)b * SEQL + l) * HEADS + h) * HD;
  float mx = -1e30f, sm = 0.f;
  for (int u = 0; u < U_SAMP; ++u) {
    int ki = hash_u(rng * 0x9E3779B9u ^ (unsigned)(l * 131 + u)) % SEQL;
    const float* kp = K + (((size_t)b * SEQL + ki) * HEADS + h) * HD;
    float d = 0.f;
    for (int j = 0; j < HD; ++j) d += q[j] * kp[j];
    mx = fmaxf(mx, d); sm += d;
  }
  Msc[idx] = mx - sm * (1.f / U_SAMP);               // layout [(b*H+h)*L + l]
}

__global__ __launch_bounds__(256) void topk_kernel(const float* __restrict__ Msc,
                                                   int* __restrict__ top)
{
  __shared__ float vals[SEQL];
  __shared__ float rv[256];
  __shared__ int ri[256];
  const int tid = threadIdx.x;
  const int bh = blockIdx.x;
  vals[tid] = Msc[(size_t)bh * SEQL + tid];
  vals[tid + 256] = Msc[(size_t)bh * SEQL + tid + 256];
  __syncthreads();
  for (int it = 0; it < U_SAMP; ++it) {
    float bv = vals[tid]; int bi = tid;
    if (vals[tid + 256] > bv) { bv = vals[tid + 256]; bi = tid + 256; }
    rv[tid] = bv; ri[tid] = bi;
    __syncthreads();
    for (int s = 128; s > 0; s >>= 1) {
      if (tid < s) {
        if (rv[tid + s] > rv[tid] ||
            (rv[tid + s] == rv[tid] && ri[tid + s] < ri[tid])) {
          rv[tid] = rv[tid + s]; ri[tid] = ri[tid + s];
        }
      }
      __syncthreads();
    }
    if (tid == 0) { top[bh * U_SAMP + it] = ri[0]; vals[ri[0]] = -1e30f; }
    __syncthreads();
  }
}

__global__ __launch_bounds__(128) void prob_scores_kernel(
    const float* __restrict__ Q, const float* __restrict__ K,
    const int* __restrict__ top, float* __restrict__ sc, int causal)
{
  int row = blockIdx.x;                              // bh*U + i
  int bh = row / U_SAMP;
  int b = bh / HEADS, h = bh % HEADS;
  int lq = top[row];
  const float* q = Q + (((size_t)b * SEQL + lq) * HEADS + h) * HD;
  for (int k = threadIdx.x; k < SEQL; k += 128) {
    const float* kp = K + (((size_t)b * SEQL + k) * HEADS + h) * HD;
    float d = 0.f;
    for (int j = 0; j < HD; ++j) d += q[j] * kp[j];
    d *= 0.125f;                                     // 1/sqrt(64)
    if (causal && k > lq) d = -1e30f;
    sc[(size_t)row * SEQL + k] = d;
  }
}

__global__ __launch_bounds__(256) void softmax_rows_kernel(float* __restrict__ x, int S)
{
  __shared__ float red[256];
  const int tid = threadIdx.x;
  float* xr = x + (size_t)blockIdx.x * S;
  float mx = -1e30f;
  for (int i = tid; i < S; i += 256) mx = fmaxf(mx, xr[i]);
  red[tid] = mx; __syncthreads();
  for (int s = 128; s > 0; s >>= 1) {
    if (tid < s) red[tid] = fmaxf(red[tid], red[tid + s]);
    __syncthreads();
  }
  mx = red[0]; __syncthreads();
  float sum = 0.f;
  for (int i = tid; i < S; i += 256) { float e = expf(xr[i] - mx); xr[i] = e; sum += e; }
  red[tid] = sum; __syncthreads();
  for (int s = 128; s > 0; s >>= 1) {
    if (tid < s) red[tid] += red[tid + s];
    __syncthreads();
  }
  float inv = 1.f / red[0];
  for (int i = tid; i < S; i += 256) xr[i] *= inv;
}

// ctx [B,H,L,HD] := cumsum_l(V) (causal) or broadcast mean_l(V)
__global__ void ctx_init_kernel(const float* __restrict__ V, float* __restrict__ ctx,
                                int causal)
{
  int idx = blockIdx.x * 256 + threadIdx.x;          // (b,h,d)
  if (idx >= BATCHN * HEADS * HD) return;
  int d = idx % HD; int h = (idx / HD) % HEADS; int b = idx / (HD * HEADS);
  const float* vcol = V + (((size_t)b * SEQL) * HEADS + h) * HD + d;   // l-stride 512
  float* cc = ctx + ((size_t)(b * HEADS + h) * SEQL) * HD + d;          // l-stride 64
  if (causal) {
    float run = 0.f;
    for (int l = 0; l < SEQL; ++l) { run += vcol[(size_t)l * DM]; cc[(size_t)l * HD] = run; }
  } else {
    float s = 0.f;
    for (int l = 0; l < SEQL; ++l) s += vcol[(size_t)l * DM];
    s *= (1.f / SEQL);
    for (int l = 0; l < SEQL; ++l) cc[(size_t)l * HD] = s;
  }
}

// ctx[b,h,top[i],:] = attn_row @ V
__global__ __launch_bounds__(64) void scatter_update_kernel(
    const float* __restrict__ attn, const float* __restrict__ V,
    const int* __restrict__ top, float* __restrict__ ctx)
{
  int row = blockIdx.x;
  int bh = row / U_SAMP;
  int b = bh / HEADS, h = bh % HEADS;
  int lt = top[row];
  int d = threadIdx.x;
  const float* ar = attn + (size_t)row * SEQL;
  float acc = 0.f;
  for (int k = 0; k < SEQL; ++k)
    acc += ar[k] * V[(((size_t)b * SEQL + k) * HEADS + h) * HD + d];
  ctx[((size_t)bh * SEQL + lt) * HD + d] = acc;
}

// mix=0: out[b,l,h,d]=ctx[b,h,l,d]; mix=1: raw flat reinterpret (Informer "mix")
__global__ void ctx_to_x_kernel(const float* __restrict__ ctx, float* __restrict__ out,
                                int mix)
{
  size_t idx = (size_t)blockIdx.x * 256 + threadIdx.x;
  if (idx >= T) return;
  if (mix) { out[idx] = ctx[idx]; return; }
  int d = idx % HD; int h = (idx / HD) % HEADS;
  int l = (idx / (HD * HEADS)) % SEQL; int b = idx / ((size_t)HD * HEADS * SEQL);
  out[idx] = ctx[(((size_t)b * HEADS + h) * SEQL + l) * HD + d];
}

// enc_ml = concat([enc_mix, dec_out], seq axis) -> [B, 2L, DM]
__global__ void concat_kernel(const float* __restrict__ a, const float* __restrict__ b2,
                              float* __restrict__ out)
{
  size_t idx = (size_t)blockIdx.x * 256 + threadIdx.x;
  if (idx >= 2 * T) return;
  int d = idx % DM; int r = (idx / DM) % (2 * SEQL); int b = idx / ((size_t)DM * 2 * SEQL);
  out[idx] = (r < SEQL) ? a[((size_t)b * SEQL + r) * DM + d]
                        : b2[((size_t)b * SEQL + (r - SEQL)) * DM + d];
}

// out[b,c] = sum_f act(X[b,f]) * W[f,c] + bias[c]   (tall-skinny projections)
__global__ __launch_bounds__(256) void proj_kernel(
    const float* __restrict__ X, const float* __restrict__ W,
    const float* __restrict__ bias, float* __restrict__ out,
    int F, int C, int relu_in)
{
  __shared__ float red[256];
  const int tid = threadIdx.x;
  const int c = blockIdx.x, b = blockIdx.y;
  const float* xb = X + (size_t)b * F;
  float acc = 0.f;
  for (int f = tid; f < F; f += 256) {
    float xv = xb[f];
    if (relu_in) xv = fmaxf(xv, 0.f);
    acc += xv * W[(size_t)f * C + c];
  }
  red[tid] = acc; __syncthreads();
  for (int s = 128; s > 0; s >>= 1) {
    if (tid < s) red[tid] += red[tid + s];
    __syncthreads();
  }
  if (tid == 0) out[b * C + c] = red[0] + (bias ? bias[c] : 0.f);
}

// ======================================================================
// Host orchestration
// ======================================================================
struct AttnP { const float *Wk,*Wo,*Wq,*Wv,*bk,*bo,*bq,*bv; };
struct EncLayerP { const float *W1,*W2; AttnP attn; const float *b1,*b2,*n1b,*n1g,*n2b,*n2g; };
struct EncP { EncLayerP lyr[3]; const float *nb,*ng; };
struct DecLayerP { const float *W1,*W2,*b1,*b2; AttnP cross; const float *n1b,*n1g,*n2b,*n2g,*n3b,*n3g; AttnP self_; };
struct DecP { DecLayerP lyr[2]; const float *nb,*ng; };

struct Cursor {
  void* const* din; int idx;
  const float* flat; size_t off; bool packed;
  const float* next(size_t n) {
    if (packed) { const float* p = flat + off; off += n; return p; }
    return (const float*)din[idx++];
  }
};
static void load_attn(Cursor& c, AttnP& a) {
  const size_t DD = (size_t)DM * DM;
  a.Wk=c.next(DD); a.Wo=c.next(DD); a.Wq=c.next(DD); a.Wv=c.next(DD);
  a.bk=c.next(DM); a.bo=c.next(DM); a.bq=c.next(DM); a.bv=c.next(DM);
}
static void load_dec(Cursor& c, DecP& d) {
  const size_t DD = (size_t)DM * DM;
  for (int i = 0; i < 2; ++i) {
    DecLayerP& y = d.lyr[i];
    y.W1=c.next(DD); y.W2=c.next(DD); y.b1=c.next(DFF); y.b2=c.next(DM);
    load_attn(c, y.cross);
    y.n1b=c.next(DM); y.n1g=c.next(DM); y.n2b=c.next(DM); y.n2g=c.next(DM);
    y.n3b=c.next(DM); y.n3g=c.next(DM);
    load_attn(c, y.self_);
  }
  d.nb=c.next(DM); d.ng=c.next(DM);
}
static void load_enc(Cursor& c, EncP& e) {
  const size_t DD = (size_t)DM * DM;
  for (int i = 0; i < 3; ++i) {
    EncLayerP& y = e.lyr[i];
    y.W1=c.next(DD); y.W2=c.next(DD);
    load_attn(c, y.attn);
    y.b1=c.next(DFF); y.b2=c.next(DM);
    y.n1b=c.next(DM); y.n1g=c.next(DM); y.n2b=c.next(DM); y.n2g=c.next(DM);
  }
  e.nb=c.next(DM); e.ng=c.next(DM);
}

extern "C" void kernel_launch(void* const* d_in, const int* in_sizes, int n_in,
                              void* d_out, int out_size, void* d_ws, size_t ws_size,
                              hipStream_t stream)
{
  (void)in_sizes; (void)out_size; (void)ws_size;
  const float* x_enc = (const float*)d_in[0];
  const float* x_dec = (const float*)d_in[1];
  float* out = (float*)d_out;

  // ---- params: leaves in jax pytree order (dict keys sorted), or one packed blob
  Cursor c;
  c.din = d_in; c.idx = 6; c.off = 0;
  c.packed = (n_in < 200);
  c.flat = c.packed ? (const float*)d_in[6] : nullptr;

  const float *Waiz = c.next(FDIM*3), *Warr = c.next(FDIM*3), *Wdepth = c.next(FDIM),
              *Wlat = c.next(FDIM*3), *Wml  = c.next(FDIM);
  const float *baiz = c.next(3), *bdepth = c.next(1), *blat = c.next(3);
  DecP dec, dec_ml; load_dec(c, dec); load_dec(c, dec_ml);
  const float *emb_dec_ml = c.next(512*9), *emb_enc = c.next(512*9);
  EncP enc_arr_p, enc_depth_p, enc_loc_p;
  load_enc(c, enc_arr_p); load_enc(c, enc_depth_p); load_enc(c, enc_loc_p);

  // ---- workspace arena (floats); requires ~350 MB
  float* ws = (float*)d_ws;
  float* enc_out  = ws + 0 * T;
  float* enc_arr  = ws + 1 * T;
  float* enc_loc  = ws + 2 * T;
  float* enc_mix  = ws + 3 * T;
  float* dec_out  = ws + 4 * T;
  float* enc_dep  = ws + 5 * T;
  float* enc_mlb  = ws + 6 * T;   // 2T
  float* dml_x    = ws + 8 * T;
  float* xbuf     = ws + 9 * T;
  float* qb       = ws + 10 * T;
  float* kb       = ws + 11 * T;  // 2T
  float* vb       = ws + 13 * T;  // 2T
  float* atmp     = ws + 15 * T;
  float* newx     = ws + 16 * T;
  float* ffy      = ws + 17 * T;
  float* ctxb     = ws + 18 * T;
  float* pscore   = ws + 19 * T;                         // BH*U*512 = 2.29M
  float* Msc      = ws + 19 * T + 3u * 1024 * 1024;      // 65536
  int*   topb     = (int*)(ws + 19 * T + 3u * 1024 * 1024 + 65536);

  auto gemm = [&](const float* A, const float* B, const float* bias, float* C,
                  int M, int act) {
    dim3 grid(GN / 64, M / 128);
    gemm_wmma_kernel<<<grid, 256, 0, stream>>>(A, B, bias, C, M, act);
  };
  auto ln = [&](const float* x, const float* res, const float* g_, const float* b_,
                float* o, int rows) {
    ln_kernel<<<rows, 256, 0, stream>>>(x, res, g_, b_, o);
  };
  auto attn_prob = [&](const AttnP& p, const float* xin, float* o, int causal,
                       int mixf, unsigned rng) {
    gemm(xin, p.Wq, p.bq, qb, BLD, 0);
    gemm(xin, p.Wk, p.bk, kb, BLD, 0);
    gemm(xin, p.Wv, p.bv, vb, BLD, 0);
    prob_sample_kernel<<<(BATCHN*HEADS*SEQL + 255) / 256, 256, 0, stream>>>(qb, kb, Msc, rng);
    topk_kernel<<<BH, 256, 0, stream>>>(Msc, topb);
    prob_scores_kernel<<<BH * U_SAMP, 128, 0, stream>>>(qb, kb, topb, pscore, causal);
    softmax_rows_kernel<<<BH * U_SAMP, 256, 0, stream>>>(pscore, SEQL);
    ctx_init_kernel<<<(BATCHN*HEADS*HD + 255) / 256, 256, 0, stream>>>(vb, ctxb, causal);
    scatter_update_kernel<<<BH * U_SAMP, HD, 0, stream>>>(pscore, vb, topb, ctxb);
    ctx_to_x_kernel<<<(int)(T / 256), 256, 0, stream>>>(ctxb, atmp, mixf);
    gemm(atmp, p.Wo, p.bo, o, BLD, 0);
  };
  auto attn_full = [&](const AttnP& p, const float* xin, const float* cross, int S,
                       float* o) {
    gemm(xin,   p.Wq, p.bq, qb, BLD, 0);
    gemm(cross, p.Wk, p.bk, kb, BATCHN * S, 0);
    gemm(cross, p.Wv, p.bv, vb, BATCHN * S, 0);
    attn_fused_kernel<<<dim3(SEQL / 64, BH), 256, 0, stream>>>(qb, kb, vb, atmp, S, 0.125f);
    gemm(atmp, p.Wo, p.bo, o, BLD, 0);
  };
  auto enc_layer = [&](const EncLayerP& p, float* x, unsigned rng) {
    attn_prob(p.attn, x, newx, 0, 0, rng);
    ln(x, newx, p.n1g, p.n1b, x, BLD);
    gemm(x, p.W1, p.b1, ffy, BLD, 1);          // fused GELU
    gemm(ffy, p.W2, p.b2, newx, BLD, 0);
    ln(x, newx, p.n2g, p.n2b, x, BLD);
  };
  auto run_encoder = [&](const EncP& e, const float* xin, float* dst, unsigned base) {
    hipMemcpyAsync(xbuf, xin, T * sizeof(float), hipMemcpyDeviceToDevice, stream);
    for (int i = 0; i < 3; ++i) enc_layer(e.lyr[i], xbuf, base + i);
    ln(xbuf, nullptr, e.ng, e.nb, dst, BLD);
  };
  auto dec_layer = [&](const DecLayerP& p, float* x, const float* cross, int S,
                       unsigned rng) {
    attn_prob(p.self_, x, newx, 1, 1, rng);
    ln(x, newx, p.n1g, p.n1b, x, BLD);
    attn_full(p.cross, x, cross, S, newx);
    ln(x, newx, p.n2g, p.n2b, x, BLD);
    gemm(x, p.W1, p.b1, ffy, BLD, 1);
    gemm(ffy, p.W2, p.b2, newx, BLD, 0);
    ln(x, newx, p.n3g, p.n3b, x, BLD);
  };
  auto run_decoder = [&](const DecP& dp, const float* xin, const float* cross, int S,
                         float* dst, unsigned base) {
    hipMemcpyAsync(xbuf, xin, T * sizeof(float), hipMemcpyDeviceToDevice, stream);
    for (int i = 0; i < 2; ++i) dec_layer(dp.lyr[i], xbuf, cross, S, base + i);
    ln(xbuf, nullptr, dp.ng, dp.nb, dst, BLD);
  };

  // ------------------- forward graph -------------------
  embed_kernel<<<(int)(T / 256), 256, 0, stream>>>(x_enc, emb_enc, enc_out);

  run_encoder(enc_arr_p, enc_out, enc_arr, 0);
  run_encoder(enc_loc_p, enc_out, enc_loc, 100);
  mix_softmax_kernel<<<(BATCHN * DM + 255) / 256, 256, 0, stream>>>(enc_arr, enc_loc, enc_mix);
  run_decoder(dec, enc_loc, enc_mix, SEQL, dec_out, 200);

  // dec_out_arrive = relu(enc_arr).reshape @ Warr        -> out[0:48]
  proj_kernel<<<dim3(3, BATCHN), 256, 0, stream>>>(enc_arr, Warr, nullptr, out + 0,  (int)FDIM, 3, 1);
  // dec_lat = pro @ Wlat + blat                          -> out[48:96]
  proj_kernel<<<dim3(3, BATCHN), 256, 0, stream>>>(dec_out, Wlat, blat,    out + 48, (int)FDIM, 3, 0);
  // dec_aiz = pro @ Waiz + baiz                          -> out[96:144]
  proj_kernel<<<dim3(3, BATCHN), 256, 0, stream>>>(dec_out, Waiz, baiz,    out + 96, (int)FDIM, 3, 0);

  run_encoder(enc_depth_p, enc_mix, enc_dep, 300);
  // dec_depth = enc_depth.reshape @ Wdepth + bdepth      -> out[144:160]
  proj_kernel<<<dim3(1, BATCHN), 256, 0, stream>>>(enc_dep, Wdepth, bdepth, out + 144, (int)FDIM, 1, 0);

  concat_kernel<<<(int)(2 * T / 256), 256, 0, stream>>>(enc_mix, dec_out, enc_mlb);
  embed_kernel<<<(int)(T / 256), 256, 0, stream>>>(x_dec, emb_dec_ml, dml_x);
  run_decoder(dec_ml, dml_x, enc_mlb, 2 * SEQL, dml_x, 400);
  // dec_out_ml = dec_ml.reshape @ Wml                    -> out[160:176]
  proj_kernel<<<dim3(1, BATCHN), 256, 0, stream>>>(dml_x, Wml, nullptr, out + 160, (int)FDIM, 1, 0);
}
